// DSS_30013231464845
// MI455X (gfx1250) — matrix operand order repaired and checked by reading, not compiled, not run on hardware
//
#include <hip/hip_runtime.h>
#include <hip/hip_bf16.h>

// Problem constants (match reference setup_inputs)
#define B_  4
#define L_  4096
#define H_  1024
#define N_  64
#define T_  64          // chunk length
#define NC_ 64          // L_/T_  number of chunks

typedef __attribute__((ext_vector_type(16))) _Float16 v16h;
typedef __attribute__((ext_vector_type(8)))  float    v8f;

__device__ __forceinline__ v8f wmma16x16x32(v16h a, v16h b, v8f c) {
  // D = A(16x32 f16) x B(32x16 f16) + C(16x16 f32)
  return __builtin_amdgcn_wmma_f32_16x16x32_f16(false, a, false, b, (short)0, c,
                                                false, false);
}

// ---- CDNA5 wave32 WMMA fragment loaders (ISA 7.12.2 layouts) ----
// A-matrix 16x32 f16: lane m=lane&15, half=lane>>4;
//   a[i]: v=i/2, sub=i&1 -> K = 16*(v/4) + 8*half + 2*(v&3) + sub
__device__ __forceinline__ v16h load_a_f16(const _Float16* base, int row0,
                                           int k0, int ld, int lane) {
  int half = lane >> 4, m = lane & 15;
  v16h a;
#pragma unroll
  for (int i = 0; i < 16; ++i) {
    int v = i >> 1, sub = i & 1;
    int k = ((v >> 2) << 4) + (half << 3) + ((v & 3) << 1) + sub;
    a[i] = base[(size_t)(row0 + m) * ld + (k0 + k)];
  }
  return a;
}
// B-matrix 32x16: lane n=lane&15, half=lane>>4; b[i] -> K = 16*(i/8) + 8*half + (i&7)
__device__ __forceinline__ v16h load_b_lds(const _Float16* base, int k0,
                                           int col0, int ld, int lane) {
  int half = lane >> 4, nn = lane & 15;
  v16h b;
#pragma unroll
  for (int i = 0; i < 16; ++i) {
    int k = ((i >> 3) << 4) + (half << 3) + (i & 7);
    b[i] = base[(k0 + k) * ld + (col0 + nn)];
  }
  return b;
}

// ================= K0: LayerNorm over H =================
__global__ void k_layernorm(const float* __restrict__ x,
                            const float* __restrict__ gamma,
                            const float* __restrict__ beta,
                            float* __restrict__ xn) {
  __shared__ float red[256];
  const int row = blockIdx.x;                 // b*L + l
  const float* xr = x + (size_t)row * H_;
  float vals[4];
  float s = 0.f;
#pragma unroll
  for (int i = 0; i < 4; ++i) { vals[i] = xr[threadIdx.x + i * 256]; s += vals[i]; }
  red[threadIdx.x] = s; __syncthreads();
  for (int off = 128; off > 0; off >>= 1) {
    if (threadIdx.x < off) red[threadIdx.x] += red[threadIdx.x + off];
    __syncthreads();
  }
  const float mu = red[0] * (1.0f / H_);
  __syncthreads();
  float v = 0.f;
#pragma unroll
  for (int i = 0; i < 4; ++i) { float d = vals[i] - mu; v += d * d; }
  red[threadIdx.x] = v; __syncthreads();
  for (int off = 128; off > 0; off >>= 1) {
    if (threadIdx.x < off) red[threadIdx.x] += red[threadIdx.x + off];
    __syncthreads();
  }
  const float rs = rsqrtf(red[0] * (1.0f / H_) + 1e-5f);
#pragma unroll
  for (int i = 0; i < 4; ++i) {
    int h = threadIdx.x + i * 256;
    xn[(size_t)row * H_ + h] = (vals[i] - mu) * rs * gamma[h] + beta[h];
  }
}

// ========= KC1: per-n constants, power tables (1 block, 64 thr) =========
__global__ void k_consts1(const float* __restrict__ LR, const float* __restrict__ LI,
                          float* Fr, float* Fi, float* Apr, float* Api,
                          float* ATr, float* ATi,
                          _Float16* Qr, _Float16* Qi, _Float16* Wcat) {
  int n = threadIdx.x;
  if (n >= N_) return;
  float lamR = -expf(LR[n]);
  float lamI =  expf(LI[n]);
  float e = expf(lamR);
  float Ar = e * cosf(lamI), Ai = e * sinf(lamI);      // A = exp(Lambda)
  float d2 = lamR * lamR + lamI * lamI;
  float nr = Ar - 1.0f, ni = Ai;                       // (exp(L)-1)/L
  Fr[n] = (nr * lamR + ni * lamI) / d2;
  Fi[n] = (ni * lamR - nr * lamI) / d2;
  float pr = 1.f, pi = 0.f;                            // A^0
  for (int t = 0; t < T_; ++t) {
    Apr[t * N_ + n] = pr; Api[t * N_ + n] = pi;        // A^t (for Ktab)
    float qr = pr * Ar - pi * Ai;
    float qi = pr * Ai + pi * Ar;
    pr = qr; pi = qi;                                  // A^{t+1}
    Wcat[t * (2 * N_) + n]       = (_Float16)pr;       // [Wr | -Wi]
    Wcat[t * (2 * N_) + N_ + n]  = (_Float16)(-pi);
  }
  ATr[n] = pr; ATi[n] = pi;                            // A^T
  for (int t = 0; t < T_; ++t) {                       // Q[n,t] = A^{T-1-t}
    int ep = T_ - 1 - t;
    Qr[n * T_ + t] = (_Float16)Apr[ep * N_ + n];
    Qi[n * T_ + t] = (_Float16)Api[ep * N_ + n];
  }
}

// ========= KC2: C[h,n] = (CR+iCI) * F[n]  =========
__global__ void k_consts2(const float* __restrict__ CRin, const float* __restrict__ CIin,
                          const float* __restrict__ Fr, const float* __restrict__ Fi,
                          float* Cr, float* Ci) {
  int idx = blockIdx.x * 256 + threadIdx.x;            // h*N + n
  int n = idx & (N_ - 1);
  float cr = CRin[idx], ci = CIin[idx];
  float fr = Fr[n], fi = Fi[n];
  Cr[idx] = cr * fr - ci * fi;
  Ci[idx] = cr * fi + ci * fr;
}

// ========= KC3: Ktab[t,h] = Re(sum_n C[h,n] A^t) =========
__global__ void k_consts3(const float* __restrict__ Cr, const float* __restrict__ Ci,
                          const float* __restrict__ Apr, const float* __restrict__ Api,
                          float* Ktab) {
  int idx = blockIdx.x * 256 + threadIdx.x;            // t*H + h
  int t = idx >> 10, h = idx & (H_ - 1);
  float s = 0.f;
#pragma unroll 8
  for (int n = 0; n < N_; ++n)
    s += Cr[h * N_ + n] * Apr[t * N_ + n] - Ci[h * N_ + n] * Api[t * N_ + n];
  Ktab[idx] = s;
}

// ========= K2: chunk-local end states  Z[b,c,:,:] = Q @ U_chunk  (WMMA) =========
__global__ void __launch_bounds__(256)
k_chunk_states(const float* __restrict__ xn,
               const _Float16* __restrict__ Qr, const _Float16* __restrict__ Qi,
               float* __restrict__ Zr, float* __restrict__ Zi) {
  __shared__ _Float16 Ush[T_ * 128];   // [t][h_local] f16, 16 KB
  const int hb = blockIdx.x;     // H/128 blocks
  const int c  = blockIdx.y;     // chunk
  const int b  = blockIdx.z;

  // --- coalesced stage of the U chunk slice (fp32 -> f16) ---
  const float* U = xn + ((size_t)b * L_ + (size_t)c * T_) * H_ + hb * 128;
  for (int idx = threadIdx.x * 4; idx < T_ * 128; idx += 256 * 4) {
    int t = idx >> 7, hl = idx & 127;
    const float4 v4 = *(const float4*)(U + (size_t)t * H_ + hl);
    Ush[idx + 0] = (_Float16)v4.x; Ush[idx + 1] = (_Float16)v4.y;
    Ush[idx + 2] = (_Float16)v4.z; Ush[idx + 3] = (_Float16)v4.w;
  }
  __syncthreads();

  const int wave = threadIdx.x >> 5, lane = threadIdx.x & 31;
  const int hl0 = wave * 16;                           // this wave's h-tile
  v16h bfrag[2];                                       // K = 64 -> 2 k-steps
#pragma unroll
  for (int ks = 0; ks < 2; ++ks) bfrag[ks] = load_b_lds(Ush, ks * 32, hl0, 128, lane);

  const size_t zbase = (((size_t)b * NC_ + c) * N_) * H_;
  const int half = lane >> 4, nn = lane & 15;
  const int hcol = hb * 128 + hl0 + nn;
#pragma unroll
  for (int ri = 0; ri < 2; ++ri) {
    const _Float16* Q = ri ? Qi : Qr;
    float* Zo = ri ? Zi : Zr;
#pragma unroll
    for (int m = 0; m < 4; ++m) {                      // n-tiles (N=64)
      v8f acc = {};
#pragma unroll
      for (int ks = 0; ks < 2; ++ks) {
        v16h a = load_a_f16(Q, m * 16, ks * 32, T_, lane);
        acc = wmma16x16x32(a, bfrag[ks], acc);
      }
#pragma unroll
      for (int r = 0; r < 8; ++r) {
        int n_idx = m * 16 + r + 8 * half;
        Zo[zbase + (size_t)n_idx * H_ + hcol] = acc[r];
      }
    }
  }
}

// ========= K3: sequential scan over chunks; in-place Z_local -> z_prev =========
__global__ void k_scan(float* __restrict__ Zr, float* __restrict__ Zi,
                       const float* __restrict__ ATr, const float* __restrict__ ATi) {
  int idx = blockIdx.x * 256 + threadIdx.x;            // b*N*H + n*H + h
  int h = idx & (H_ - 1);
  int n = (idx >> 10) & (N_ - 1);
  int b = idx >> 16;
  const float ar = ATr[n], ai = ATi[n];
  const size_t cstride = (size_t)N_ * H_;              // chunk stride in floats
  float cr = 0.f, ci = 0.f;
  for (int c = 0; c < NC_; ++c) {
    size_t p = (((size_t)b * NC_ + c) * N_ + n) * H_ + h;
    if (c + 1 < NC_) {                                 // prefetch next chunk state
      __builtin_prefetch(&Zr[p + cstride], 0, 1);
      __builtin_prefetch(&Zi[p + cstride], 0, 1);
    }
    float tr = Zr[p], ti = Zi[p];
    Zr[p] = cr; Zi[p] = ci;                            // z_prev for chunk c
    float nr2 = ar * cr - ai * ci + tr;
    float ni2 = ar * ci + ai * cr + ti;
    cr = nr2; ci = ni2;
  }
}

// ========= K4: output = Wcat @ (C .* z_prev)  + depthwise causal conv + D*u =========
__global__ void __launch_bounds__(256)
k_output(const float* __restrict__ xn,
         const float* __restrict__ Zr, const float* __restrict__ Zi,
         const float* __restrict__ Cr, const float* __restrict__ Ci,
         const _Float16* __restrict__ Wcat,
         const float* __restrict__ Ktab, const float* __restrict__ Dv,
         float* __restrict__ y) {
  __shared__ _Float16 Gs[128 * 128];   // [k'=2N][h_local]  32 KB
  __shared__ float    Us[T_ * 128];    // [t][h_local]      32 KB
  const int hb = blockIdx.x, c = blockIdx.y, b = blockIdx.z;
  const int h0 = hb * 128;
  const int tid = threadIdx.x;

  // stage u chunk slice (fp32, coalesced float4)
  const float* U = xn + ((size_t)b * L_ + (size_t)c * T_) * H_ + h0;
  for (int idx = tid * 4; idx < T_ * 128; idx += 256 * 4) {
    int t = idx >> 7, hl = idx & 127;
    *(float4*)(Us + idx) = *(const float4*)(U + (size_t)t * H_ + hl);
  }
  // G = C .* z_prev  (complex elementwise), packed [Gr ; Gi] as f16
  const size_t zbase = (((size_t)b * NC_ + c) * N_) * H_ + h0;
  for (int idx = tid; idx < N_ * 128; idx += 256) {
    int n = idx >> 7, hl = idx & 127;
    float zr = Zr[zbase + (size_t)n * H_ + hl];
    float zi = Zi[zbase + (size_t)n * H_ + hl];
    float cr = Cr[(size_t)(h0 + hl) * N_ + n];
    float ci = Ci[(size_t)(h0 + hl) * N_ + n];
    Gs[n * 128 + hl]         = (_Float16)(cr * zr - ci * zi);
    Gs[(N_ + n) * 128 + hl]  = (_Float16)(cr * zi + ci * zr);
  }
  __syncthreads();

  const int wave = tid >> 5, lane = tid & 31;
  const int half = lane >> 4, nn = lane & 15;
  const int hl0 = wave * 16;                           // wave's h-tile in block
  const int col = hl0 + nn;                            // lane's fixed h column
  const int hglob = h0 + col;

  // ---- carry matmul: hoisted B fragments (reused across all 4 m-tiles) ----
  v16h bfr[4];
#pragma unroll
  for (int ks = 0; ks < 4; ++ks) bfr[ks] = load_b_lds(Gs, ks * 32, hl0, 128, lane);

  v8f accv[4];
#pragma unroll
  for (int m = 0; m < 4; ++m) {                        // lc-tiles (T=64)
    v8f acc = {};
#pragma unroll
    for (int ks = 0; ks < 4; ++ks) {                   // K' = 2N = 128
      v16h a = load_a_f16(Wcat, m * 16, ks * 32, 2 * N_, lane);
      acc = wmma16x16x32(a, bfr[ks], acc);
    }
    accv[m] = acc;
  }

  // ---- registerize the lane's Ktab column (statically indexed) ----
  float Kt[T_];
#pragma unroll
  for (int t = 0; t < T_; ++t) Kt[t] = Ktab[(size_t)t * H_ + hglob];

  // ---- intra-chunk causal conv, fully static triangular unroll ----
  // lane output lc = m*16 + r + 8*half; fold 8*half rows into the base address.
  const float* UsL = Us + (half << 3) * 128 + col;     // &Us[8*half][col]
#pragma unroll
  for (int t = 0; t < T_; ++t) {
#pragma unroll
    for (int m = 0; m < 4; ++m) {
#pragma unroll
      for (int r = 0; r < 8; ++r) {
        const int d0 = m * 16 + r - t;                 // static
        if (d0 >= 0) {
          // valid for both halves: row = d0 + 8*half (static imm off of UsL)
          accv[m][r] += Kt[t] * UsL[d0 * 128];
        } else if (d0 + 8 >= 0) {
          // valid only for half==1 lanes (lc = m*16+r+8 >= t)
          float kv = half ? Kt[t] : 0.0f;
          accv[m][r] += kv * Us[(d0 + 8) * 128 + col];
        }
      }
    }
  }

  // ---- skip term + store ----
  const float d = Dv[hglob];
#pragma unroll
  for (int m = 0; m < 4; ++m) {
#pragma unroll
    for (int r = 0; r < 8; ++r) {
      int lc = m * 16 + r + 8 * half;
      float uval = Us[lc * 128 + col];
      size_t oidx = (((size_t)b * L_ + (size_t)c * T_ + lc) * H_) + hglob;
      y[oidx] = accv[m][r] + d * uval;
    }
  }
}

// ========================= launcher =========================
extern "C" void kernel_launch(void* const* d_in, const int* in_sizes, int n_in,
                              void* d_out, int out_size, void* d_ws, size_t ws_size,
                              hipStream_t stream) {
  const float* x     = (const float*)d_in[0];
  const float* gamma = (const float*)d_in[1];
  const float* beta  = (const float*)d_in[2];
  const float* LR    = (const float*)d_in[3];
  const float* LI    = (const float*)d_in[4];
  const float* CR    = (const float*)d_in[5];
  const float* CI    = (const float*)d_in[6];
  const float* Dv    = (const float*)d_in[7];
  float* y = (float*)d_out;

  // workspace carve-up (~202 MB total)
  char* ws = (char*)d_ws;
  size_t off = 0;
  auto carve = [&](size_t bytes) -> void* {
    off = (off + 255) & ~(size_t)255;
    void* p = ws + off; off += bytes; return p;
  };
  const size_t nBLH = (size_t)B_ * L_ * H_;
  float*    xn   = (float*)carve(nBLH * 4);
  float*    Zr   = (float*)carve(nBLH * 4);   // [B][NC][N][H] (NC*N == L)
  float*    Zi   = (float*)carve(nBLH * 4);
  float*    Apr  = (float*)carve((size_t)T_ * N_ * 4);
  float*    Api  = (float*)carve((size_t)T_ * N_ * 4);
  float*    Fr   = (float*)carve(N_ * 4);
  float*    Fi   = (float*)carve(N_ * 4);
  float*    ATr  = (float*)carve(N_ * 4);
  float*    ATi  = (float*)carve(N_ * 4);
  float*    Crw  = (float*)carve((size_t)H_ * N_ * 4);
  float*    Ciw  = (float*)carve((size_t)H_ * N_ * 4);
  float*    Ktab = (float*)carve((size_t)T_ * H_ * 4);
  _Float16* Qr   = (_Float16*)carve((size_t)N_ * T_ * 2);
  _Float16* Qi   = (_Float16*)carve((size_t)N_ * T_ * 2);
  _Float16* Wcat = (_Float16*)carve((size_t)T_ * 2 * N_ * 2);
  (void)ws_size; (void)in_sizes; (void)n_in; (void)out_size;

  // 1) LayerNorm
  k_layernorm<<<B_ * L_, 256, 0, stream>>>(x, gamma, beta, xn);
  // 2) constants
  k_consts1<<<1, 64, 0, stream>>>(LR, LI, Fr, Fi, Apr, Api, ATr, ATi, Qr, Qi, Wcat);
  k_consts2<<<(H_ * N_) / 256, 256, 0, stream>>>(CR, CI, Fr, Fi, Crw, Ciw);
  k_consts3<<<(T_ * H_) / 256, 256, 0, stream>>>(Crw, Ciw, Apr, Api, Ktab);
  // 3) chunk-local states (WMMA)
  k_chunk_states<<<dim3(H_ / 128, NC_, B_), 256, 0, stream>>>(xn, Qr, Qi, Zr, Zi);
  // 4) cross-chunk scan (with next-chunk prefetch)
  k_scan<<<(B_ * N_ * H_) / 256, 256, 0, stream>>>(Zr, Zi, ATr, ATi);
  // 5) carry matmul (WMMA) + intra-chunk conv + skip
  k_output<<<dim3(H_ / 128, NC_, B_), 256, 0, stream>>>(xn, Zr, Zi, Crw, Ciw,
                                                        Wcat, Ktab, Dv, y);
}
